// Head_1872605741926
// MI455X (gfx1250) — matrix operand order repaired and checked by reading, not compiled, not run on hardware
//
#include <hip/hip_runtime.h>

#define BATCH 8
#define SEQ   4096
#define CDIM  64
#define HD    64

typedef _Float16 h8   __attribute__((ext_vector_type(8)));
typedef _Float16 h16  __attribute__((ext_vector_type(16)));
typedef float    f8v  __attribute__((ext_vector_type(8)));

union H16 { h16 v; h8 h[2]; };

// ---------------------------------------------------------------------------
// Projection: q = (x Wq^T) * C^-0.5, k = x Wk^T, v = x Wv^T (f32 math, f16 out).
// The attention scale 1/8 is folded into q. vT is stored transposed
// (vT[b][h][t]) so flash-kernel V tiles arrive h-major.
// ---------------------------------------------------------------------------
__global__ __launch_bounds__(256) void proj_kernel(
    const float* __restrict__ x,
    const float* __restrict__ Wk,
    const float* __restrict__ Wq,
    const float* __restrict__ Wv,
    _Float16* __restrict__ qf,
    _Float16* __restrict__ kf,
    _Float16* __restrict__ vTf)
{
    __shared__ float4 sWq[64 * 16];
    __shared__ float4 sWk[64 * 16];
    __shared__ float4 sWv[64 * 16];

    const int tid = threadIdx.x;
    for (int i = tid; i < 64 * 16; i += 256) {
        sWq[i] = ((const float4*)Wq)[i];
        sWk[i] = ((const float4*)Wk)[i];
        sWv[i] = ((const float4*)Wv)[i];
    }
    __syncthreads();

    const int g = blockIdx.x * 256 + tid;          // global token index
    const int b = g / SEQ;
    const int t = g % SEQ;

    float4 xr[16];
    const float4* xp = (const float4*)(x + (size_t)g * CDIM);
#pragma unroll
    for (int i = 0; i < 16; ++i) xr[i] = xp[i];

    _Float16* qo = qf + (size_t)g * HD;
    _Float16* ko = kf + (size_t)g * HD;

    for (int h = 0; h < HD; ++h) {
        float aq = 0.f, ak = 0.f, av = 0.f;
#pragma unroll
        for (int c = 0; c < 16; ++c) {
            float4 xv = xr[c];
            float4 wq = sWq[h * 16 + c];
            float4 wk = sWk[h * 16 + c];
            float4 wv = sWv[h * 16 + c];
            aq += xv.x * wq.x + xv.y * wq.y + xv.z * wq.z + xv.w * wq.w;
            ak += xv.x * wk.x + xv.y * wk.y + xv.z * wk.z + xv.w * wk.w;
            av += xv.x * wv.x + xv.y * wv.y + xv.z * wv.z + xv.w * wv.w;
        }
        qo[h] = (_Float16)(aq * 0.125f);   // fold C^-0.5 = 64^-0.5 into q
        ko[h] = (_Float16)ak;
        vTf[((size_t)(b * HD + h)) * SEQ + t] = (_Float16)av;
    }
}

// ---------------------------------------------------------------------------
// Flash attention, causal, head dim 64.
// Block: 256 threads = 8 waves; each wave owns 16 query rows (128 rows/block).
// Key loop: 64 keys/step, double-buffered async-to-LDS prefetch (DMA for
// tile i+1 overlaps the 18 v_wmma + softmax of tile i).
// ---------------------------------------------------------------------------
#define BC        64   // keys per iteration
#define KT_STRIDE 72   // halves per K-tile row (144 B: 16B aligned, bank-spread)
#define VT_STRIDE 72   // halves per V-tile row
#define PS_STRIDE 64   // halves per P-scratch row

__device__ __forceinline__ void async_ld_b128(unsigned ldsByteOff,
                                              unsigned memByteOff,
                                              const void* base)
{
    asm volatile("global_load_async_to_lds_b128 %0, %1, %2"
                 :
                 : "v"(ldsByteOff), "v"(memByteOff),
                   "s"((unsigned long long)(uintptr_t)base)
                 : "memory");
}

__global__ __launch_bounds__(256) void flash_kernel(
    const _Float16* __restrict__ qf,
    const _Float16* __restrict__ kf,
    const _Float16* __restrict__ vTf,
    float* __restrict__ out)
{
    __shared__ __attribute__((aligned(16))) _Float16 kTile[2][BC * KT_STRIDE]; // [key][d]
    __shared__ __attribute__((aligned(16))) _Float16 vTile[2][HD * VT_STRIDE]; // [h][key]
    __shared__ __attribute__((aligned(16))) _Float16 pS[8 * 16 * PS_STRIDE];

    const int tid  = threadIdx.x;
    const int wave = tid >> 5;
    const int lane = tid & 31;
    const int hi   = lane >> 4;    // half-lane group
    const int ln   = lane & 15;

    const int tilesPerB = SEQ / 128;
    const int b     = blockIdx.x / tilesPerB;
    const int row0  = (blockIdx.x % tilesPerB) * 128;
    const int qbase = row0 + wave * 16;
    const int qend  = qbase + 15;
    const int kEnd  = row0 + 128;  // exclusive key bound for this block

    // per-thread staging chunks (2 x 16B per array per tile)
    int sKey[2], sD[2];
#pragma unroll
    for (int i = 0; i < 2; ++i) {
        int chunk = tid + 256 * i;
        sKey[i] = chunk >> 3;
        sD[i]   = (chunk & 7) * 8;
    }

    // --- load (pre-scaled) Q as two 16x32 A-fragments ---
    H16 qa[2];
    const _Float16* qrow = qf + ((size_t)(b * SEQ + qbase + ln)) * HD;
#pragma unroll
    for (int c = 0; c < 2; ++c) {
        qa[c].h[0] = *(const h8*)(qrow + 32 * c + 16 * hi);
        qa[c].h[1] = *(const h8*)(qrow + 32 * c + 16 * hi + 8);
    }

    // B fragment of all-ones: P x 1s -> per-row running sum in o[4]
    h16 onesv;
#pragma unroll
    for (int i = 0; i < 16; ++i) onesv[i] = (_Float16)1.0f;

    f8v o[5];                      // o[0..3] = output head chunks, o[4] = row sums
    float m[8];
#pragma unroll
    for (int j = 0; j < 8; ++j) {
        m[j] = -1e30f;
#pragma unroll
        for (int f = 0; f < 5; ++f) o[f][j] = 0.f;
    }

    _Float16* ps = pS + wave * (16 * PS_STRIDE);

    // --- prologue: issue async loads for tile 0 into buffer 0 ---
#pragma unroll
    for (int i = 0; i < 2; ++i) {
        async_ld_b128((unsigned)(uintptr_t)&kTile[0][sKey[i] * KT_STRIDE + sD[i]],
                      (unsigned)(((b * SEQ + sKey[i]) * HD + sD[i]) * 2), kf);
        async_ld_b128((unsigned)(uintptr_t)&vTile[0][sKey[i] * VT_STRIDE + sD[i]],
                      (unsigned)(((b * HD + sKey[i]) * SEQ + sD[i]) * 2), vTf);
    }

    for (int kb = 0; kb < kEnd; kb += BC) {
        const int cur = (kb >> 6) & 1;
        const int nxt = cur ^ 1;

        __syncthreads();   // all waves done reading buffer `nxt` (2 tiles ago)

        const bool more = (kb + BC) < kEnd;
        if (more) {
            const int kb2 = kb + BC;
#pragma unroll
            for (int i = 0; i < 2; ++i) {
                async_ld_b128((unsigned)(uintptr_t)&kTile[nxt][sKey[i] * KT_STRIDE + sD[i]],
                              (unsigned)(((b * SEQ + kb2 + sKey[i]) * HD + sD[i]) * 2), kf);
                async_ld_b128((unsigned)(uintptr_t)&vTile[nxt][sKey[i] * VT_STRIDE + sD[i]],
                              (unsigned)(((b * HD + sKey[i]) * SEQ + (kb2 + sD[i])) * 2), vTf);
            }
            // async loads complete in order: <=4 outstanding means current
            // tile's 4 loads landed; next tile's 4 may still be in flight.
            asm volatile("s_wait_asynccnt 0x4" ::: "memory");
        } else {
            asm volatile("s_wait_asynccnt 0x0" ::: "memory");
        }
        __syncthreads();   // every wave's current-tile loads are in LDS

        if (kb > qend) continue;   // wave-uniform; barriers stay at loop top

        // --- S = Q K^T : four 16x16 score fragments (keys kb+16t) ---
        f8v s[4];
#pragma unroll
        for (int t = 0; t < 4; ++t)
#pragma unroll
            for (int j = 0; j < 8; ++j) s[t][j] = 0.f;
#pragma unroll
        for (int c = 0; c < 2; ++c) {
#pragma unroll
            for (int t = 0; t < 4; ++t) {
                // B fragment: lane holds key = 16t + ln, contiguous d = 32c+16hi..+15
                h16 bk = *(const h16*)&kTile[cur][(16 * t + ln) * KT_STRIDE + 32 * c + 16 * hi];
                s[t] = __builtin_amdgcn_wmma_f32_16x16x32_f16(false, qa[c].v, false, bk,
                                                              (short)0, s[t], false, false);
            }
        }

        const bool diag = (kb + BC - 1 > qbase);   // any masking needed in this tile?
#pragma unroll
        for (int j = 0; j < 8; ++j) {
            const int row = qbase + j + 8 * hi;    // D layout: M = j + 8*hi
            float vv[4];
#pragma unroll
            for (int t = 0; t < 4; ++t) vv[t] = s[t][j];   // scale pre-folded into Q
            if (diag) {
#pragma unroll
                for (int t = 0; t < 4; ++t)
                    if (kb + 16 * t + ln > row) vv[t] = -1e30f;
            }
            // row max over 64 columns: 4-way local + butterfly over 16-lane half
            float mx = fmaxf(fmaxf(vv[0], vv[1]), fmaxf(vv[2], vv[3]));
#pragma unroll
            for (int off = 1; off < 16; off <<= 1)
                mx = fmaxf(mx, __shfl_xor(mx, off, 16));
            const float mnew  = fmaxf(m[j], mx);
            const float alpha = __expf(m[j] - mnew);
            m[j] = mnew;
#pragma unroll
            for (int f = 0; f < 5; ++f) o[f][j] *= alpha;  // rescale O and row sums
            // stash P (f16) for D-layout -> A-layout transpose via LDS
#pragma unroll
            for (int t = 0; t < 4; ++t)
                ps[(j + 8 * hi) * PS_STRIDE + 16 * t + ln] =
                    (_Float16)__expf(vv[t] - mnew);
        }

        // Re-read P as two 16x32 A-fragments (wave-private LDS; DS ops in-order)
        H16 pa[2];
#pragma unroll
        for (int kc = 0; kc < 2; ++kc) {
            pa[kc].h[0] = *(const h8*)(ps + ln * PS_STRIDE + 32 * kc + 16 * hi);
            pa[kc].h[1] = *(const h8*)(ps + ln * PS_STRIDE + 32 * kc + 16 * hi + 8);
        }

        // --- O += P V (4 head chunks x 2 key chunks), row sums += P x 1s ---
#pragma unroll
        for (int kc = 0; kc < 2; ++kc) {
#pragma unroll
            for (int hc = 0; hc < 4; ++hc) {
                // B fragment: lane holds h = 16hc + ln, keys 32kc + 16hi + (0..15)
                h16 bv = *(const h16*)&vTile[cur][(16 * hc + ln) * VT_STRIDE + 32 * kc + 16 * hi];
                o[hc] = __builtin_amdgcn_wmma_f32_16x16x32_f16(false, pa[kc].v, false, bv,
                                                               (short)0, o[hc], false, false);
            }
            o[4] = __builtin_amdgcn_wmma_f32_16x16x32_f16(false, pa[kc].v, false, onesv,
                                                          (short)0, o[4], false, false);
        }
    }

    // --- normalize and write out (f32, [B][T][HD]); o[4] holds row sums l ---
#pragma unroll
    for (int hc = 0; hc < 4; ++hc) {
#pragma unroll
        for (int j = 0; j < 8; ++j) {
            const int row = qbase + j + 8 * hi;
            out[((size_t)(b * SEQ) + row) * HD + 16 * hc + ln] = o[hc][j] / o[4][j];
        }
    }
}

// ---------------------------------------------------------------------------
extern "C" void kernel_launch(void* const* d_in, const int* in_sizes, int n_in,
                              void* d_out, int out_size, void* d_ws, size_t ws_size,
                              hipStream_t stream) {
    (void)in_sizes; (void)n_in; (void)out_size; (void)ws_size;
    const float* x  = (const float*)d_in[0];
    const float* Wk = (const float*)d_in[1];
    const float* Wq = (const float*)d_in[2];
    const float* Wv = (const float*)d_in[3];

    char* ws = (char*)d_ws;
    const size_t nElem = (size_t)BATCH * SEQ * HD;     // 2,097,152
    _Float16* qf  = (_Float16*)(ws);
    _Float16* kf  = (_Float16*)(ws + nElem * 2);
    _Float16* vTf = (_Float16*)(ws + nElem * 4);

    proj_kernel<<<BATCH * SEQ / 256, 256, 0, stream>>>(x, Wk, Wq, Wv, qf, kf, vTf);
    flash_kernel<<<BATCH * (SEQ / 128), 256, 0, stream>>>(qf, kf, vTf, (float*)d_out);
}
